// MultiHeadAttention_28845000359981
// MI455X (gfx1250) — matrix operand order repaired and checked by reading, not compiled, not run on hardware
//
#include <hip/hip_runtime.h>
#include <hip/hip_bf16.h>

#define D_MODEL 768
#define NHEAD   12
#define DK      64
#define SEQ     4096
#define BATCH   2
#define MROWS   (BATCH * SEQ)   // 8192
#define KBLK    64              // keys staged per block iteration
#define KSTRIDE 72              // padded LDS row stride (halves) for K/V tiles
#define PSTRIDE 24              // padded LDS row stride (halves) for P^T

typedef _Float16 h8  __attribute__((ext_vector_type(8)));
typedef _Float16 h16 __attribute__((ext_vector_type(16)));
typedef float    f8  __attribute__((ext_vector_type(8)));

__device__ __forceinline__ f8 wmma_f16(h16 a, h16 b, f8 c) {
  // (neg_a, A, neg_b, B, c_mod, C, reuse_a, reuse_b)
  return __builtin_amdgcn_wmma_f32_16x16x32_f16(false, a, false, b, (short)0, c,
                                                false, false);
}

__device__ __forceinline__ h16 cat16(h8 lo, h8 hi) {
  h16 r;
#pragma unroll
  for (int i = 0; i < 8; ++i) { r[i] = lo[i]; r[8 + i] = hi[i]; }
  return r;
}

// gfx1250 async copy: global -> LDS, 16B per lane, tracked by ASYNCcnt.
// VDST operand = LDS byte offset (low 32 bits of a generic LDS pointer).
__device__ __forceinline__ void async_load_b128(unsigned lds_off,
                                                const _Float16* gaddr) {
  asm volatile("global_load_async_to_lds_b128 %0, %1, off"
               :: "v"(lds_off), "v"(gaddr)
               : "memory");
}

__device__ __forceinline__ void wait_async0() {
  asm volatile("s_wait_asynccnt 0x0" ::: "memory");
}

// ---------------------------------------------------------------------------
// Kernel 1: W[o][i] (f32) -> W^T[i][o] (f16), for Wq,Wk,Wv,Wo.
// B-operand of WMMA wants lane = K-row with N contiguous, i.e. [in][out].
// ---------------------------------------------------------------------------
__global__ void wtrans_kernel(const float* __restrict__ Wq,
                              const float* __restrict__ Wk,
                              const float* __restrict__ Wv,
                              const float* __restrict__ Wo,
                              _Float16* __restrict__ wt) {
  int mat = blockIdx.y;
  const float* W = (mat == 0) ? Wq : (mat == 1) ? Wk : (mat == 2) ? Wv : Wo;
  _Float16* dst = wt + (size_t)mat * D_MODEL * D_MODEL;
  int idx = blockIdx.x * blockDim.x + threadIdx.x;   // 0 .. 768*768-1
  int o = idx % D_MODEL;                             // contiguous across lanes
  int i = idx / D_MODEL;
  dst[(size_t)i * D_MODEL + o] = (_Float16)W[(size_t)o * D_MODEL + i];
}

// ---------------------------------------------------------------------------
// Kernel 2: fused QKV projection.  GEMM M=8192, N=768, K=768 per matrix.
// Q   stored [b][h][s][dk]  (A-operand layout for QK^T)
// K^T stored [b][h][dk][s]  (B-operand layout for QK^T)
// V^T stored [b][h][dk][s]  (A-operand layout for ctx^T = V^T P^T)
// One wave computes a 16(M) x 64(N) tile; 4 WMMA per k-step of 32.
// ---------------------------------------------------------------------------
__global__ void qkv_kernel(const float* __restrict__ x,
                           const _Float16* __restrict__ wt,
                           const float* __restrict__ bq,
                           const float* __restrict__ bk,
                           const float* __restrict__ bv,
                           _Float16* __restrict__ qb,
                           _Float16* __restrict__ ktb,
                           _Float16* __restrict__ vtb) {
  int lane = threadIdx.x & 31, w = threadIdx.x >> 5;
  int l16 = lane & 15, grp = lane >> 4;
  int r0 = (blockIdx.x * 4 + w) * 16;
  int h = blockIdx.y;               // head == 64-wide N tile
  int n0 = h * DK;
  int mat = blockIdx.z;             // 0=Q 1=K 2=V

  const _Float16* W = wt + (size_t)mat * D_MODEL * D_MODEL;
  const float* bias = (mat == 0) ? bq : (mat == 1) ? bk : bv;

  f8 acc[4] = {};
  const float* xrow = x + (size_t)(r0 + l16) * D_MODEL;

  for (int kk = 0; kk < D_MODEL / 32; ++kk) {
    int d0 = kk * 32 + grp * 8;                 // A chunk base (16-bit A layout)
    f8 x0 = *(const f8*)(xrow + d0);
    f8 x1 = *(const f8*)(xrow + d0 + 16);
    h16 a;
#pragma unroll
    for (int i = 0; i < 8; ++i) { a[i] = (_Float16)x0[i]; a[8 + i] = (_Float16)x1[i]; }

    const _Float16* wrow = W + (size_t)(kk * 32 + grp * 8) * D_MODEL + n0 + l16;
#pragma unroll
    for (int j = 0; j < 4; ++j) {
      h16 bm;
#pragma unroll
      for (int i = 0; i < 8; ++i) {
        bm[i] = wrow[(size_t)i * D_MODEL + j * 16];
        bm[8 + i] = wrow[(size_t)(16 + i) * D_MODEL + j * 16];
      }
      acc[j] = wmma_f16(a, bm, acc[j]);
    }
  }

  if (mat == 0) {   // Q: row-major per head, scattered b16 stores
#pragma unroll
    for (int j = 0; j < 4; ++j) {
      int dk = j * 16 + l16;
      float bi = bias[n0 + dk];
#pragma unroll
      for (int v = 0; v < 8; ++v) {
        int r = r0 + v + 8 * grp;
        int bb = r >> 12, sl = r & (SEQ - 1);
        qb[(((size_t)bb * NHEAD + h) * SEQ + sl) * DK + dk] = (_Float16)(acc[j][v] + bi);
      }
    }
  } else {          // K^T / V^T: rows contiguous along VGPR index -> b128 stores
    _Float16* dst = (mat == 1) ? ktb : vtb;
    int r = r0 + 8 * grp;
    int bb = r >> 12, sl = r & (SEQ - 1);
#pragma unroll
    for (int j = 0; j < 4; ++j) {
      int dk = j * 16 + l16;
      float bi = bias[n0 + dk];
      h8 st;
#pragma unroll
      for (int v = 0; v < 8; ++v) st[v] = (_Float16)(acc[j][v] + bi);
      *(h8*)(dst + (((size_t)bb * NHEAD + h) * DK + dk) * SEQ + sl) = st;
    }
  }
}

// ---------------------------------------------------------------------------
// Kernel 3: flash attention.  Block = 4 waves, one (b,h), 64 query rows.
// Per 64-key block: K/V tiles (64x64 f16, padded rows) staged into LDS with
// global_load_async_to_lds_b128 (shared by all 4 waves), 8 WMMA for
// S = Q K^T, online softmax (shfl reductions in 16-lane halves, base-2 with
// 1/8*log2e folded into Q), P^T through per-wave LDS into B-operand layout,
// 8 WMMA for ctx^T += V^T P^T.
// ---------------------------------------------------------------------------
__global__ void flash_kernel(const _Float16* __restrict__ qb,
                             const _Float16* __restrict__ ktb,
                             const _Float16* __restrict__ vtb,
                             _Float16* __restrict__ ctx) {
  __shared__ __align__(16) _Float16 Ktile[DK * KSTRIDE];       // 9216 B
  __shared__ __align__(16) _Float16 Vtile[DK * KSTRIDE];       // 9216 B
  __shared__ __align__(16) _Float16 Pbuf[4][KBLK * PSTRIDE];   // 12 KB
  __shared__ float Abuf[4][16];

  int tid = threadIdx.x;
  int lane = tid & 31, w = tid >> 5;
  int l16 = lane & 15, grp = lane >> 4;
  int sq0 = (blockIdx.x * 4 + w) * 16;     // query tile base
  int h = blockIdx.y, b = blockIdx.z;
  size_t bh = (size_t)b * NHEAD + h;

  const _Float16* Q  = qb  + bh * (size_t)SEQ * DK;
  const _Float16* KT = ktb + bh * (size_t)DK * SEQ;
  const _Float16* VT = vtb + bh * (size_t)DK * SEQ;
  _Float16* Pw = Pbuf[w];
  float*    Aw = Abuf[w];
  const unsigned kbase = (unsigned)(size_t)(void*)Ktile;   // LDS byte offsets
  const unsigned vbase = (unsigned)(size_t)(void*)Vtile;

  const float SCALE = 0.125f * 1.44269504088896340736f;  // 1/sqrt(64) * log2(e)

  // Q A-operands (K=32 each), pre-scaled so softmax runs in base 2.
  h16 qa[2];
  {
    const _Float16* qrow = Q + (size_t)(sq0 + l16) * DK;
#pragma unroll
    for (int kk = 0; kk < 2; ++kk) {
      int d0 = kk * 32 + grp * 8;
      h16 a = cat16(*(const h8*)(qrow + d0), *(const h8*)(qrow + d0 + 16));
#pragma unroll
      for (int i = 0; i < 16; ++i) a[i] = (_Float16)((float)a[i] * SCALE);
      qa[kk] = a;
    }
  }

  f8 acc[4] = {};
  float m_run[8], l_run[8];
#pragma unroll
  for (int v = 0; v < 8; ++v) { m_run[v] = -__builtin_inff(); l_run[v] = 0.f; }

  for (int t0 = 0; t0 < SEQ; t0 += KBLK) {
    // Stage K/V 64x64 tiles: 512 x 16B per matrix, 4 per thread per matrix.
#pragma unroll
    for (int k = 0; k < 4; ++k) {
      int i = tid + 128 * k;                 // 0..511
      int d = i >> 3, c = (i & 7) * 8;       // row, half-offset within row
      unsigned loff = (unsigned)((d * KSTRIDE + c) * 2);
      async_load_b128(kbase + loff, KT + (size_t)d * SEQ + t0 + c);
      async_load_b128(vbase + loff, VT + (size_t)d * SEQ + t0 + c);
    }
    wait_async0();
    __syncthreads();   // tiles written by all waves now visible

    // S tiles (D-layout: M=s in VGPR idx, N=t in lane), 8 WMMA
    f8 sacc[4] = {};
#pragma unroll
    for (int kk = 0; kk < 2; ++kk) {
      const _Float16* kr = Ktile + (kk * 32 + grp * 8) * KSTRIDE + l16;
#pragma unroll
      for (int j = 0; j < 4; ++j) {
        h16 kb;
#pragma unroll
        for (int i = 0; i < 8; ++i) {
          kb[i] = kr[i * KSTRIDE + j * 16];
          kb[8 + i] = kr[(16 + i) * KSTRIDE + j * 16];
        }
        sacc[j] = wmma_f16(qa[kk], kb, sacc[j]);
      }
    }

    // Online softmax over 64 keys; P^T written straight into per-wave LDS.
    float alpha[8];
#pragma unroll
    for (int v = 0; v < 8; ++v) {
      float mv = fmaxf(fmaxf(sacc[0][v], sacc[1][v]),
                       fmaxf(sacc[2][v], sacc[3][v]));
#pragma unroll
      for (int msk = 1; msk < 16; msk <<= 1) mv = fmaxf(mv, __shfl_xor(mv, msk, 32));
      float mn = fmaxf(m_run[v], mv);
      alpha[v] = __builtin_amdgcn_exp2f(m_run[v] - mn);
      float rs = 0.f;
#pragma unroll
      for (int j = 0; j < 4; ++j) {
        float p = __builtin_amdgcn_exp2f(sacc[j][v] - mn);
        Pw[(j * 16 + l16) * PSTRIDE + v + 8 * grp] = (_Float16)p;
        rs += p;
      }
#pragma unroll
      for (int msk = 1; msk < 16; msk <<= 1) rs += __shfl_xor(rs, msk, 32);
      l_run[v] = l_run[v] * alpha[v] + rs;
      m_run[v] = mn;
    }
    if (l16 == 0) {
#pragma unroll
      for (int v = 0; v < 8; ++v) Aw[8 * grp + v] = alpha[v];
    }
    // P/alpha are per-wave; LDS ops are in-order within a wave -> no barrier.

    float asf = Aw[l16];   // rescale per-lane (N=s in ctx^T layout)
#pragma unroll
    for (int mt = 0; mt < 4; ++mt)
#pragma unroll
      for (int v = 0; v < 8; ++v) acc[mt][v] *= asf;

#pragma unroll
    for (int kkp = 0; kkp < 2; ++kkp) {      // K-dim = keys, 32 per WMMA
      const _Float16* pr = Pw + (kkp * 32 + grp * 8) * PSTRIDE + l16;
      h16 pb;
#pragma unroll
      for (int i = 0; i < 8; ++i) { pb[i] = pr[i * PSTRIDE]; pb[8 + i] = pr[(16 + i) * PSTRIDE]; }
#pragma unroll
      for (int mt = 0; mt < 4; ++mt) {
        const _Float16* vr = Vtile + (mt * 16 + l16) * KSTRIDE + kkp * 32 + grp * 8;
        acc[mt] = wmma_f16(cat16(*(const h8*)vr, *(const h8*)(vr + 16)), pb, acc[mt]);
      }
    }
    __syncthreads();   // all waves done reading K/V tiles before next stage
  }

  if (l16 == 0) {
#pragma unroll
    for (int v = 0; v < 8; ++v) Aw[8 * grp + v] = l_run[v];
  }
  __syncthreads();
  float linv = 1.0f / Aw[l16];

  // ctx^T D-layout: d along VGPR idx -> contiguous b128 f16 stores into [b][s][D]
  size_t row = (size_t)b * SEQ + sq0 + l16;
#pragma unroll
  for (int mt = 0; mt < 4; ++mt) {
    h8 st;
#pragma unroll
    for (int v = 0; v < 8; ++v) st[v] = (_Float16)(acc[mt][v] * linv);
    *(h8*)(ctx + row * D_MODEL + h * DK + mt * 16 + grp * 8) = st;
  }
}

// ---------------------------------------------------------------------------
// Kernel 4: out = ctx @ Wo^T + bo (f32 output).  M=8192, N=768, K=768.
// ---------------------------------------------------------------------------
__global__ void oproj_kernel(const _Float16* __restrict__ ctx,
                             const _Float16* __restrict__ wto,
                             const float* __restrict__ bo,
                             float* __restrict__ out) {
  int lane = threadIdx.x & 31, w = threadIdx.x >> 5;
  int l16 = lane & 15, grp = lane >> 4;
  int r0 = (blockIdx.x * 4 + w) * 16;
  int n0 = blockIdx.y * 64;

  f8 acc[4] = {};
  const _Float16* arow = ctx + (size_t)(r0 + l16) * D_MODEL;
  for (int kk = 0; kk < D_MODEL / 32; ++kk) {
    int d0 = kk * 32 + grp * 8;
    h16 a = cat16(*(const h8*)(arow + d0), *(const h8*)(arow + d0 + 16));
    const _Float16* wr = wto + (size_t)(kk * 32 + grp * 8) * D_MODEL + n0 + l16;
#pragma unroll
    for (int j = 0; j < 4; ++j) {
      h16 bm;
#pragma unroll
      for (int i = 0; i < 8; ++i) {
        bm[i] = wr[(size_t)i * D_MODEL + j * 16];
        bm[8 + i] = wr[(size_t)(16 + i) * D_MODEL + j * 16];
      }
      acc[j] = wmma_f16(a, bm, acc[j]);
    }
  }
#pragma unroll
  for (int j = 0; j < 4; ++j) {
    int n = n0 + j * 16 + l16;
    float bi = bo[n];
#pragma unroll
    for (int v = 0; v < 8; ++v)
      out[(size_t)(r0 + v + 8 * grp) * D_MODEL + n] = acc[j][v] + bi;
  }
}

// ---------------------------------------------------------------------------
extern "C" void kernel_launch(void* const* d_in, const int* in_sizes, int n_in,
                              void* d_out, int out_size, void* d_ws, size_t ws_size,
                              hipStream_t stream) {
  (void)in_sizes; (void)n_in; (void)out_size; (void)ws_size;
  const float* x  = (const float*)d_in[0];
  const float* Wq = (const float*)d_in[1];
  const float* bq = (const float*)d_in[2];
  const float* Wk = (const float*)d_in[3];
  const float* bk = (const float*)d_in[4];
  const float* Wv = (const float*)d_in[5];
  const float* bv = (const float*)d_in[6];
  const float* Wo = (const float*)d_in[7];
  const float* bo = (const float*)d_in[8];
  float* out = (float*)d_out;

  const size_t WMAT = (size_t)D_MODEL * D_MODEL;           // halves per matrix
  const size_t HBUF = (size_t)BATCH * NHEAD * SEQ * DK;    // halves per Q/K/V buf
  _Float16* wt  = (_Float16*)d_ws;          // 4 transposed f16 weight matrices
  _Float16* qb  = wt  + 4 * WMAT;           // Q   [b][h][s][dk]
  _Float16* ktb = qb  + HBUF;               // K^T [b][h][dk][s]
  _Float16* vtb = ktb + HBUF;               // V^T [b][h][dk][s]
  _Float16* ctx = vtb + HBUF;               // ctx [b][s][D]

  wtrans_kernel<<<dim3(D_MODEL * D_MODEL / 256, 4), 256, 0, stream>>>(Wq, Wk, Wv, Wo, wt);
  qkv_kernel<<<dim3(MROWS / 64, NHEAD, 3), 128, 0, stream>>>(x, wt, bq, bk, bv, qb, ktb, vtb);
  flash_kernel<<<dim3(SEQ / 64, NHEAD, BATCH), 128, 0, stream>>>(qb, ktb, vtb, ctx);
  oproj_kernel<<<dim3(MROWS / 64, D_MODEL / 64), 128, 0, stream>>>(ctx, wt + 3 * WMAT, bo, out);
}
